// GraphAttentionLayer_88837103550733
// MI455X (gfx1250) — matrix operand order repaired and checked by reading, not compiled, not run on hardware
//
#include <hip/hip_runtime.h>
#include <math.h>

// ---------------------------------------------------------------------------
// GraphAttentionLayer on MI455X (gfx1250, wave32, WMMA).
//
// Roofline: ~90 GFLOP total vs ~130 MB of HBM traffic @ 23.3 TB/s -> compute
// bound; run all matmuls through v_wmma_f32_16x16x32_bf16 (f32 accumulate).
// All transposes are hoisted out of inner loops:
//   * weights converted to bf16 pre-transposed [n][k]  -> B-fragments are
//     contiguous, GEMM stages B with one ds_store_b128/thread,
//   * V written [b,h,d,s] by its projection GEMM        -> attention P@V
//     B-fragments stream straight from global (L2-resident), so the flash
//     attention inner loop has NO LDS staging and NO barriers at all.
// ---------------------------------------------------------------------------

typedef __bf16 bf16_t;
typedef __attribute__((ext_vector_type(8)))  __bf16 bf16x8;
typedef __attribute__((ext_vector_type(16))) __bf16 bf16x16;
typedef __attribute__((ext_vector_type(8)))  float  f32x8;

union BF16x16 { bf16x16 v; bf16x8 h[2]; };

#define WMMA_BF16(a, b, c) \
  __builtin_amdgcn_wmma_f32_16x16x32_bf16(false, (a), false, (b), (short)0, (c), false, false)

// Problem constants (fixed by the reference).
#define PB 4
#define PS 2048
#define PH 768
#define PNH 6
#define PD 128

// ---------------------------------------------------------------------------
// f32 -> bf16 elementwise convert
// ---------------------------------------------------------------------------
__global__ void cvt_f32_to_bf16(const float* __restrict__ src,
                                bf16_t* __restrict__ dst, int n) {
  int i = blockIdx.x * blockDim.x + threadIdx.x;
  if (i < n) dst[i] = (bf16_t)src[i];
}

// f32 [K][N] -> bf16 transposed [N][K]  (weights; done once, 0.6M elems each)
__global__ void cvt_w_transpose_bf16(const float* __restrict__ src,
                                     bf16_t* __restrict__ dst) {
  int i = blockIdx.x * blockDim.x + threadIdx.x;  // coalesced read
  if (i < PH * PH) {
    int k = i / PH, n = i % PH;
    dst[(size_t)n * PH + k] = (bf16_t)src[i];
  }
}

// ---------------------------------------------------------------------------
// Tiled bf16 GEMM with WMMA:  Y = (X[M,K] @ W[K,N] + bias) * scale
// Wt is PRE-TRANSPOSED: Wt[n][k]. Block: 256 threads = 8 waves.
// Tile: 128(M) x 64(N), BK=32. A-fragments straight from global (contiguous);
// B tile staged in LDS with one ds_store_b128 per thread (block-level reuse).
// MODE 0: bf16 out scattered to [b, head, s, d]   (Q, K)
// MODE 2: bf16 out scattered to [b, head, d, s]   (V, pre-transposed for P@V)
// MODE 1: f32 out, row-major [M, N]               (final projection)
// ---------------------------------------------------------------------------
template <int MODE>
__global__ __launch_bounds__(256)
void gemm_bf16_wmma(const bf16_t* __restrict__ X, const bf16_t* __restrict__ Wt,
                    const float* __restrict__ bias, void* __restrict__ outp,
                    int M, int N, int K, float scale) {
  constexpr int BM = 128, BN = 64, BK = 32;
  __shared__ bf16_t Bs[BN * BK];  // [n][k], 4 KB

  const int t    = threadIdx.x;
  const int wv   = t >> 5;
  const int lane = t & 31;
  const int lo   = lane & 15;   // WMMA m / n index
  const int hi   = lane >> 4;   // WMMA k-group select
  const int m0   = blockIdx.y * BM;
  const int n0   = blockIdx.x * BN;
  const bf16_t* arow = X + (size_t)(m0 + wv * 16 + lo) * K;

  f32x8 acc[4] = {};

  for (int k0 = 0; k0 < K; k0 += BK) {
    // Stage B tile: Wt is already [n][k], so one 16B vector store per thread.
    {
      int n = t >> 2, ko = (t & 3) * 8;
      *(bf16x8*)&Bs[n * BK + ko] =
          *(const bf16x8*)&Wt[(size_t)(n0 + n) * K + k0 + ko];
    }
    __syncthreads();

    // A fragment from global: lane holds row (m0+wv*16+lo), k = hi*16+{0..15}.
    BF16x16 a;
    a.h[0] = *(const bf16x8*)&arow[k0 + hi * 16];
    a.h[1] = *(const bf16x8*)&arow[k0 + hi * 16 + 8];
#pragma unroll
    for (int nb = 0; nb < 4; ++nb) {
      // B fragment: lane holds col (nb*16+lo), 16 contiguous k at hi*16.
      bf16x16 b = *(const bf16x16*)&Bs[(nb * 16 + lo) * BK + hi * 16];
      acc[nb] = WMMA_BF16(a.v, b, acc[nb]);
    }
    __syncthreads();
  }

  // Epilogue. C layout: VGPR r -> row r + 8*hi, col lo.
#pragma unroll
  for (int nb = 0; nb < 4; ++nb) {
    int n_g = n0 + nb * 16 + lo;
    float bv = bias[n_g];
#pragma unroll
    for (int r = 0; r < 8; ++r) {
      int m_g = m0 + wv * 16 + r + 8 * hi;
      float v = (acc[nb][r] + bv) * scale;
      int bb = m_g >> 11;          // / PS
      int ss = m_g & (PS - 1);
      int head = n_g >> 7;         // / PD
      int dc = n_g & (PD - 1);
      if (MODE == 0) {
        ((bf16_t*)outp)[(((size_t)bb * PNH + head) * PS + ss) * PD + dc] = (bf16_t)v;
      } else if (MODE == 2) {
        ((bf16_t*)outp)[(((size_t)bb * PNH + head) * PD + dc) * PS + ss] = (bf16_t)v;
      } else {
        ((float*)outp)[(size_t)m_g * N + n_g] = v;
      }
    }
  }
}

// ---------------------------------------------------------------------------
// Flash attention, bf16 WMMA, masked online softmax. Barrier-free inner loop.
// Grid: (S/128, B*NH). Block: 256 threads = 8 waves; each wave owns 16 queries.
// Q,K: [b,h,s,d] bf16 (Q pre-scaled by 1/sqrt(d) incl. bias). V: [b,h,d,s].
// ---------------------------------------------------------------------------
__global__ __launch_bounds__(256)
void flash_attn_bf16(const bf16_t* __restrict__ Q, const bf16_t* __restrict__ Km,
                     const bf16_t* __restrict__ Vt,
                     const unsigned char* __restrict__ mask,  // bool [B,S,S]
                     bf16_t* __restrict__ ctx) {              // [B,S,H] bf16
  constexpr int PSTR = 32;
  __shared__ bf16_t Psh[8 * 16 * PSTR];  // per-wave P tiles 16x32 -> 8 KB

  const int t    = threadIdx.x;
  const int wv   = t >> 5;
  const int lane = t & 31;
  const int lo   = lane & 15;
  const int hi   = lane >> 4;
  const int bh   = blockIdx.y;
  const int b    = bh / PNH;
  const int head = bh % PNH;
  const int q0   = blockIdx.x * 128 + wv * 16;

  const bf16_t* Qh = Q  + (size_t)bh * PS * PD;   // [s][d]
  const bf16_t* Kh = Km + (size_t)bh * PS * PD;   // [s][d]
  const bf16_t* Vh = Vt + (size_t)bh * PD * PS;   // [d][s]  (pre-transposed)
  bf16_t* Pl = &Psh[wv * 16 * PSTR];

  // Persistent Q fragments: 4 chunks of 16x32 (A layout), read once.
  BF16x16 qf[4];
  {
    const bf16_t* qrow = Qh + (size_t)(q0 + lo) * PD;
#pragma unroll
    for (int c = 0; c < 4; ++c) {
      qf[c].h[0] = *(const bf16x8*)&qrow[c * 32 + hi * 16];
      qf[c].h[1] = *(const bf16x8*)&qrow[c * 32 + hi * 16 + 8];
    }
  }

  f32x8 oacc[8] = {};
  float mrow[8], lrow[8];
#pragma unroll
  for (int r = 0; r < 8; ++r) { mrow[r] = -3.0e38f; lrow[r] = 0.0f; }

  for (int j = 0; j < PS / 32; ++j) {
    const int key0 = j * 32;

    // Prefetch next key block's K rows and V lines into L2/L0.
    if (j + 1 < PS / 32) {
      const bf16_t* pk = Kh + (size_t)(key0 + 32 + lane) * PD;
      __builtin_prefetch(pk, 0, 3);
      __builtin_prefetch(pk + 64, 0, 3);
      const bf16_t* pv = Vh + (size_t)(lane * 4) * PS + key0 + 32;
#pragma unroll
      for (int i = 0; i < 4; ++i) __builtin_prefetch(pv + (size_t)i * PS, 0, 3);
    }

    // Scores for 32 keys: K read from global in B-fragment layout
    // (16 contiguous dims of key (key0 [+16] + lo) at dim-chunk hi*16).
    f32x8 s0 = {}, s1 = {};
#pragma unroll
    for (int c = 0; c < 4; ++c) {
      const bf16_t* kb = Kh + (size_t)key0 * PD + c * 32 + hi * 16;
      bf16x16 kf0 = *(const bf16x16*)&kb[(size_t)lo * PD];
      bf16x16 kf1 = *(const bf16x16*)&kb[(size_t)(lo + 16) * PD];
      s0 = WMMA_BF16(qf[c].v, kf0, s0);
      s1 = WMMA_BF16(qf[c].v, kf1, s1);
    }

    // Mask bias + online softmax (rows live in 16-lane halves; shfl_xor with
    // masks < 16 stays inside each half on wave32).
#pragma unroll
    for (int r = 0; r < 8; ++r) {
      int rB = r + 8 * hi;
      int qg = q0 + rB;
      const unsigned char* mp = mask + ((size_t)b * PS + qg) * PS + key0;
      float x0 = s0[r] + (mp[lo]      ? 0.0f : -9.0e15f);
      float x1 = s1[r] + (mp[lo + 16] ? 0.0f : -9.0e15f);
      float mx = fmaxf(x0, x1);
#pragma unroll
      for (int off = 1; off < 16; off <<= 1)
        mx = fmaxf(mx, __shfl_xor(mx, off, 32));
      float mnew = fmaxf(mrow[r], mx);
      float p0 = __expf(x0 - mnew);
      float p1 = __expf(x1 - mnew);
      float ps = p0 + p1;
#pragma unroll
      for (int off = 1; off < 16; off <<= 1)
        ps += __shfl_xor(ps, off, 32);
      float corr = __expf(mrow[r] - mnew);
      lrow[r] = lrow[r] * corr + ps;
      mrow[r] = mnew;
#pragma unroll
      for (int nb = 0; nb < 8; ++nb) oacc[nb][r] *= corr;
      Pl[rB * PSTR + lo]      = (bf16_t)p0;  // C-layout -> LDS (intra-wave)
      Pl[rB * PSTR + lo + 16] = (bf16_t)p1;
    }

    // Reload P in A-fragment layout (same wave; LDS ops are in-order -> no barrier).
    BF16x16 pf;
    pf.h[0] = *(const bf16x8*)&Pl[lo * PSTR + hi * 16];
    pf.h[1] = *(const bf16x8*)&Pl[lo * PSTR + hi * 16 + 8];

    // O += P(16x32) @ V(32x128): V is [d][s], so B-fragments (16 contiguous
    // keys at fixed d) stream straight from global / L2.
#pragma unroll
    for (int nb = 0; nb < 8; ++nb) {
      bf16x16 vf = *(const bf16x16*)&Vh[(size_t)(nb * 16 + lo) * PS + key0 + hi * 16];
      oacc[nb] = WMMA_BF16(pf.v, vf, oacc[nb]);
    }
  }

  // Normalize and store context in [B, S, H] bf16.
#pragma unroll
  for (int nb = 0; nb < 8; ++nb) {
    int col = head * PD + nb * 16 + lo;
#pragma unroll
    for (int r = 0; r < 8; ++r) {
      int rB = r + 8 * hi;
      int qg = q0 + rB;
      float v = oacc[nb][r] / lrow[r];
      ctx[((size_t)b * PS + qg) * PH + col] = (bf16_t)v;
    }
  }
}

// ---------------------------------------------------------------------------
// Host-side launcher
// ---------------------------------------------------------------------------
extern "C" void kernel_launch(void* const* d_in, const int* in_sizes, int n_in,
                              void* d_out, int out_size, void* d_ws, size_t ws_size,
                              hipStream_t stream) {
  (void)in_sizes; (void)n_in; (void)out_size; (void)ws_size;

  const float* seq  = (const float*)d_in[0];
  const float* Wq   = (const float*)d_in[1];
  const float* bq   = (const float*)d_in[2];
  const float* Wk   = (const float*)d_in[3];
  const float* bk   = (const float*)d_in[4];
  const float* Wv   = (const float*)d_in[5];
  const float* bv   = (const float*)d_in[6];
  const float* Wf   = (const float*)d_in[7];
  const float* bfin = (const float*)d_in[8];
  const unsigned char* mask = (const unsigned char*)d_in[9];  // jnp bool = 1 byte

  const int M = PB * PS;          // 8192
  const int nX = M * PH;          // 6,291,456
  const int nW = PH * PH;         // 589,824

  // Workspace carve-out (~67.6 MB), all offsets well aligned.
  char* ws = (char*)d_ws;
  size_t off = 0;
  bf16_t* x_bf   = (bf16_t*)(ws + off); off += (size_t)nX * 2;
  bf16_t* wqt    = (bf16_t*)(ws + off); off += (size_t)nW * 2;
  bf16_t* wkt    = (bf16_t*)(ws + off); off += (size_t)nW * 2;
  bf16_t* wvt    = (bf16_t*)(ws + off); off += (size_t)nW * 2;
  bf16_t* wft    = (bf16_t*)(ws + off); off += (size_t)nW * 2;
  bf16_t* q_bf   = (bf16_t*)(ws + off); off += (size_t)nX * 2;  // [b,h,s,d]
  bf16_t* k_bf   = (bf16_t*)(ws + off); off += (size_t)nX * 2;  // [b,h,s,d]
  bf16_t* v_t    = (bf16_t*)(ws + off); off += (size_t)nX * 2;  // [b,h,d,s]
  bf16_t* ctx_bf = (bf16_t*)(ws + off); off += (size_t)nX * 2;  // [b,s,H]

  // 1) bf16 conversions; weights transposed once so inner loops never transpose.
  cvt_f32_to_bf16<<<(nX + 255) / 256, 256, 0, stream>>>(seq, x_bf, nX);
  cvt_w_transpose_bf16<<<(nW + 255) / 256, 256, 0, stream>>>(Wq, wqt);
  cvt_w_transpose_bf16<<<(nW + 255) / 256, 256, 0, stream>>>(Wk, wkt);
  cvt_w_transpose_bf16<<<(nW + 255) / 256, 256, 0, stream>>>(Wv, wvt);
  cvt_w_transpose_bf16<<<(nW + 255) / 256, 256, 0, stream>>>(Wf, wft);

  // 2) QKV projections (score scale 1/sqrt(128) folded into Q incl. bias).
  dim3 gg(PH / 64, M / 128);  // (12, 64)
  const float qscale = 0.08838834764831845f;
  gemm_bf16_wmma<0><<<gg, 256, 0, stream>>>(x_bf, wqt, bq, q_bf, M, PH, PH, qscale);
  gemm_bf16_wmma<0><<<gg, 256, 0, stream>>>(x_bf, wkt, bk, k_bf, M, PH, PH, 1.0f);
  gemm_bf16_wmma<2><<<gg, 256, 0, stream>>>(x_bf, wvt, bv, v_t,  M, PH, PH, 1.0f);

  // 3) Flash attention with graph-mask bias (barrier-free inner loop).
  dim3 ga(PS / 128, PB * PNH);  // (16, 24)
  flash_attn_bf16<<<ga, 256, 0, stream>>>(q_bf, k_bf, v_t, mask, ctx_bf);

  // 4) Final projection, f32 output.
  gemm_bf16_wmma<1><<<gg, 256, 0, stream>>>(ctx_bf, wft, bfin, (float*)d_out,
                                            M, PH, PH, 1.0f);
}
